// WindowAttention_11605001634297
// MI455X (gfx1250) — matrix operand order repaired and checked by reading, not compiled, not run on hardware
//
#include <hip/hip_runtime.h>

typedef float     v8f  __attribute__((ext_vector_type(8)));
typedef float     v4f  __attribute__((ext_vector_type(4)));
typedef _Float16  v8h  __attribute__((ext_vector_type(8)));
typedef _Float16  v16h __attribute__((ext_vector_type(16)));

#define NTOK 49      // tokens per window
#define CDIM 128     // channels
#define HEADS 4
#define HD 32        // head dim
#define MPAD 64      // padded token count (4 tiles of 16)

// LDS strides (in elements). Row-stride bytes are all multiples of 16 (b128-clean)
// and equal 4 (or 36) dwords mod 64 banks -> conflict-free lane-r row walks.
#define QKS 264      // q|k f16 buffer stride   (row 528 B)
#define VS  72       // v-transposed f16 stride (row 144 B)
#define XS2 136      // x f16 stride            (row 272 B)
#define SSF 68       // scores f32 stride       (row 272 B)
#define PS  72       // P f16 stride            (row 144 B)
#define AOS 136      // attn-out f16 stride     (row 272 B)

// LDS byte offsets (all 16B-aligned)
#define OFF_QK   0                               // 64*264*2  = 33792
#define OFF_VT   33792                           // 128*72*2  = 18432
#define OFF_X    52224                           // 64*136*2  = 17408 (dead after QKV)
#define OFF_SC   52224                           // 64*68*4   = 17408 (aliases x)
#define OFF_P    69632                           // 64*72*2   = 9216
#define OFF_AO   78848                           // 64*136*2  = 17408
#define SMEM_BYTES 96256                         // 94 KB -> 3 WGs / WGP

#define WQ_ELEMS (3 * CDIM * CDIM)               // 49152
#define WP_ELEMS (CDIM * CDIM)                   // 16384

static __device__ __forceinline__ v16h cat16(v8h lo, v8h hi) {
  return __builtin_shufflevector(lo, hi, 0, 1, 2, 3, 4, 5, 6, 7,
                                 8, 9, 10, 11, 12, 13, 14, 15);
}
static __device__ __forceinline__ v8f wmma16(v16h a, v16h b, v8f c) {
  // D(16x16 f32) = A(16x32 f16) x B(32x16 f16) + C
  return __builtin_amdgcn_wmma_f32_16x16x32_f16(false, a, false, b, (short)0, c,
                                                false, false);
}

// ---- one-shot weight conversion into workspace (f16, row-major) ----
__global__ __launch_bounds__(256) void cvt_weights_kernel(
    const float* __restrict__ qkv_w, const float* __restrict__ proj_w,
    _Float16* __restrict__ wq, _Float16* __restrict__ wp) {
  int i = blockIdx.x * 256 + threadIdx.x;
  if (i < WQ_ELEMS) wq[i] = (_Float16)qkv_w[i];
  if (i < WP_ELEMS) wp[i] = (_Float16)proj_w[i];
}

__global__ __launch_bounds__(256) void winattn_kernel(
    const float* __restrict__ x, const float* __restrict__ mask,
    const _Float16* __restrict__ wq, const float* __restrict__ qkv_b,
    const _Float16* __restrict__ wp, const float* __restrict__ proj_b,
    float* __restrict__ out) {
  extern __shared__ char smem[];
  _Float16* qk_h = (_Float16*)(smem + OFF_QK);  // [64][QKS]: cols 0-127 q*scale, 128-255 k
  _Float16* vT_h = (_Float16*)(smem + OFF_VT);  // [128][VS]: vT[dim][token]
  _Float16* x_h  = (_Float16*)(smem + OFF_X);   // [64][XS2]
  float*    sc_s = (float*)   (smem + OFF_SC);  // [64][SSF] raw scores
  _Float16* p_h  = (_Float16*)(smem + OFF_P);   // [64][PS]  softmaxed P
  _Float16* ao_h = (_Float16*)(smem + OFF_AO);  // [64][AOS]

  const int b    = blockIdx.x;
  const int tid  = threadIdx.x;
  const int wave = tid >> 5;
  const int lane = tid & 31;
  const int hf   = lane >> 4;       // wave half
  const int r    = lane & 15;
  const int ka   = 8 * hf;          // A-frag K offsets: ka, 16+ka (two 16B chunks)
  const int kb   = 8 * hf;          // B-frag K offsets: kb, 16+kb (two 16B chunks)
  const float scale = 0.17677669529663687f;  // 32^-0.5

  // ---------------- Phase 0: stage x -> f16 (rows >= 49 zero); zero P pad rows ----
  const float* xb = x + (size_t)b * (NTOK * CDIM);
  for (int idx = tid; idx < MPAD * (CDIM / 4); idx += 256) {
    int row = idx >> 5, c4 = idx & 31;
    v4f v = {0.f, 0.f, 0.f, 0.f};
    if (row < NTOK) v = *(const v4f*)(xb + row * CDIM + c4 * 4);
    _Float16* d = x_h + row * XS2 + c4 * 4;
    d[0] = (_Float16)v[0]; d[1] = (_Float16)v[1];
    d[2] = (_Float16)v[2]; d[3] = (_Float16)v[3];
  }
  for (int idx = tid; idx < (MPAD - NTOK) * PS; idx += 256)
    p_h[NTOK * PS + idx] = (_Float16)0.f;   // rows 49..63 of P stay zero forever
  __syncthreads();

  // ---------------- Phase 1: QKV = x @ qkv_w^T + qkv_b (f16 in, f32 acc) --------
  // M=64, N=384 (24 tiles), K=128 (4 steps). Wave: 3 N-tiles, B reused over 4 M-tiles.
  for (int ntq = wave; ntq < 24; ntq += 8) {
    const int colBase = ntq * 16;
    const float bias = qkv_b[colBase + r];
    v8f c0, c1, c2, c3;
#pragma unroll
    for (int i = 0; i < 8; ++i) { c0[i] = bias; c1[i] = bias; c2[i] = bias; c3[i] = bias; }
    const _Float16* wrow = wq + (size_t)(colBase + r) * CDIM;
#pragma unroll
    for (int kk = 0; kk < CDIM; kk += 32) {
      v16h bf = cat16(*(const v8h*)(wrow + kk + kb), *(const v8h*)(wrow + kk + 16 + kb));
      const _Float16* a0p = x_h + (0 * 16 + r) * XS2 + kk;
      const _Float16* a1p = x_h + (1 * 16 + r) * XS2 + kk;
      const _Float16* a2p = x_h + (2 * 16 + r) * XS2 + kk;
      const _Float16* a3p = x_h + (3 * 16 + r) * XS2 + kk;
      c0 = wmma16(cat16(*(const v8h*)(a0p + ka), *(const v8h*)(a0p + 16 + ka)), bf, c0);
      c1 = wmma16(cat16(*(const v8h*)(a1p + ka), *(const v8h*)(a1p + 16 + ka)), bf, c1);
      c2 = wmma16(cat16(*(const v8h*)(a2p + ka), *(const v8h*)(a2p + 16 + ka)), bf, c2);
      c3 = wmma16(cat16(*(const v8h*)(a3p + ka), *(const v8h*)(a3p + 16 + ka)), bf, c3);
    }
    if (colBase < 2 * CDIM) {                      // q (scaled) and k -> qk_h
      const float mul = (colBase < CDIM) ? scale : 1.0f;
#pragma unroll
      for (int i = 0; i < 8; ++i) {
        qk_h[(0 * 16 + i + 8 * hf) * QKS + colBase + r] = (_Float16)(c0[i] * mul);
        qk_h[(1 * 16 + i + 8 * hf) * QKS + colBase + r] = (_Float16)(c1[i] * mul);
        qk_h[(2 * 16 + i + 8 * hf) * QKS + colBase + r] = (_Float16)(c2[i] * mul);
        qk_h[(3 * 16 + i + 8 * hf) * QKS + colBase + r] = (_Float16)(c3[i] * mul);
      }
    } else {                                       // v -> transposed vT[dim][token]
      const int dim = colBase - 2 * CDIM + r;
#pragma unroll
      for (int i = 0; i < 8; ++i) {
        vT_h[dim * VS + 0 * 16 + i + 8 * hf] = (_Float16)c0[i];
        vT_h[dim * VS + 1 * 16 + i + 8 * hf] = (_Float16)c1[i];
        vT_h[dim * VS + 2 * 16 + i + 8 * hf] = (_Float16)c2[i];
        vT_h[dim * VS + 3 * 16 + i + 8 * hf] = (_Float16)c3[i];
      }
    }
  }
  __syncthreads();   // x_h dead; sc_s/p_h/ao_h region live

  // ---------------- Phase 2: per-head attention ----------------
  const float* mrow = mask + (size_t)(b & 63) * (NTOK * NTOK);

  for (int h = 0; h < HEADS; ++h) {
    // ---- scores = (q*scale) @ k^T : K=32 -> single wmma per tile; 2 tiles/wave
    {
      const int nt = wave & 3;
      const int mt0 = (wave >> 2) * 2;
      v8f s0, s1;
#pragma unroll
      for (int i = 0; i < 8; ++i) { s0[i] = 0.f; s1[i] = 0.f; }
      const _Float16* krow  = qk_h + (nt * 16 + r) * QKS + CDIM + h * HD;
      const _Float16* qrow0 = qk_h + ((mt0    ) * 16 + r) * QKS + h * HD;
      const _Float16* qrow1 = qk_h + ((mt0 + 1) * 16 + r) * QKS + h * HD;
      v16h bf = cat16(*(const v8h*)(krow + kb), *(const v8h*)(krow + 16 + kb));
      s0 = wmma16(cat16(*(const v8h*)(qrow0 + ka), *(const v8h*)(qrow0 + 16 + ka)), bf, s0);
      s1 = wmma16(cat16(*(const v8h*)(qrow1 + ka), *(const v8h*)(qrow1 + 16 + ka)), bf, s1);
#pragma unroll
      for (int i = 0; i < 8; ++i) {
        sc_s[((mt0    ) * 16 + i + 8 * hf) * SSF + nt * 16 + r] = s0[i];
        sc_s[((mt0 + 1) * 16 + i + 8 * hf) * SSF + nt * 16 + r] = s1[i];
      }
    }
    __syncthreads();

    // ---- masked softmax in f32 (wave per row), write P as f16
    for (int row = wave; row < NTOK; row += 8) {
      float raw0 = sc_s[row * SSF + lane];
      float raw1 = sc_s[row * SSF + lane + 32];
      float v0 = (lane < NTOK)      ? raw0 + mrow[row * NTOK + lane]      : -3.0e38f;
      float v1 = (lane + 32 < NTOK) ? raw1 + mrow[row * NTOK + lane + 32] : -3.0e38f;
      float mx = fmaxf(v0, v1);
#pragma unroll
      for (int off = 16; off > 0; off >>= 1) mx = fmaxf(mx, __shfl_xor(mx, off, 32));
      float e0 = __expf(v0 - mx);
      float e1 = __expf(v1 - mx);
      float sum = e0 + e1;
#pragma unroll
      for (int off = 16; off > 0; off >>= 1) sum += __shfl_xor(sum, off, 32);
      float inv = 1.0f / sum;
      p_h[row * PS + lane]      = (_Float16)(e0 * inv);
      p_h[row * PS + lane + 32] = (_Float16)(e1 * inv);
    }
    __syncthreads();

    // ---- attn-out = P @ v : M=64, N=32, K=64 (2 steps); 1 tile/wave
    {
      const int mt = wave >> 1;
      const int nt = wave & 1;
      v8f c;
#pragma unroll
      for (int i = 0; i < 8; ++i) c[i] = 0.f;
      const _Float16* vrow = vT_h + (h * HD + nt * 16 + r) * VS;  // B[k][n]=vT[n][k]
      const _Float16* prow = p_h + (mt * 16 + r) * PS;
#pragma unroll
      for (int kk = 0; kk < MPAD; kk += 32) {
        v16h bf = cat16(*(const v8h*)(vrow + kk + kb), *(const v8h*)(vrow + kk + 16 + kb));
        v16h af = cat16(*(const v8h*)(prow + kk + ka), *(const v8h*)(prow + kk + 16 + ka));
        c = wmma16(af, bf, c);
      }
#pragma unroll
      for (int i = 0; i < 8; ++i)
        ao_h[(mt * 16 + i + 8 * hf) * AOS + h * HD + nt * 16 + r] = (_Float16)c[i];
    }
    __syncthreads();
  }

  // ---------------- Phase 3: out = attn_out @ proj_w^T + proj_b ----------------
  {
    const int colBase = wave * 16;      // 8 N-tiles, one per wave
    const float bias = proj_b[colBase + r];
    v8f c0, c1, c2, c3;
#pragma unroll
    for (int i = 0; i < 8; ++i) { c0[i] = bias; c1[i] = bias; c2[i] = bias; c3[i] = bias; }
    const _Float16* wrow = wp + (size_t)(colBase + r) * CDIM;
#pragma unroll
    for (int kk = 0; kk < CDIM; kk += 32) {
      v16h bf = cat16(*(const v8h*)(wrow + kk + kb), *(const v8h*)(wrow + kk + 16 + kb));
      const _Float16* a0p = ao_h + (0 * 16 + r) * AOS + kk;
      const _Float16* a1p = ao_h + (1 * 16 + r) * AOS + kk;
      const _Float16* a2p = ao_h + (2 * 16 + r) * AOS + kk;
      const _Float16* a3p = ao_h + (3 * 16 + r) * AOS + kk;
      c0 = wmma16(cat16(*(const v8h*)(a0p + ka), *(const v8h*)(a0p + 16 + ka)), bf, c0);
      c1 = wmma16(cat16(*(const v8h*)(a1p + ka), *(const v8h*)(a1p + 16 + ka)), bf, c1);
      c2 = wmma16(cat16(*(const v8h*)(a2p + ka), *(const v8h*)(a2p + 16 + ka)), bf, c2);
      c3 = wmma16(cat16(*(const v8h*)(a3p + ka), *(const v8h*)(a3p + 16 + ka)), bf, c3);
    }
    float* ob = out + (size_t)b * (NTOK * CDIM);
#pragma unroll
    for (int i = 0; i < 8; ++i) {
      int m0 = 0 * 16 + i + 8 * hf;
      int m1 = 1 * 16 + i + 8 * hf;
      int m2 = 2 * 16 + i + 8 * hf;
      int m3 = 3 * 16 + i + 8 * hf;
      ob[m0 * CDIM + colBase + r] = c0[i];
      ob[m1 * CDIM + colBase + r] = c1[i];
      ob[m2 * CDIM + colBase + r] = c2[i];
      if (m3 < NTOK) ob[m3 * CDIM + colBase + r] = c3[i];
    }
  }
}

extern "C" void kernel_launch(void* const* d_in, const int* in_sizes, int n_in,
                              void* d_out, int out_size, void* d_ws, size_t ws_size,
                              hipStream_t stream) {
  (void)in_sizes; (void)n_in; (void)ws_size; (void)out_size;
  const float* x      = (const float*)d_in[0];
  const float* mask   = (const float*)d_in[1];
  const float* qkv_w  = (const float*)d_in[2];
  const float* qkv_b  = (const float*)d_in[3];
  const float* proj_w = (const float*)d_in[4];
  const float* proj_b = (const float*)d_in[5];
  float* out = (float*)d_out;

  _Float16* wq = (_Float16*)d_ws;            // 49152 f16
  _Float16* wp = wq + WQ_ELEMS;              // 16384 f16 (131072 B total)

  cvt_weights_kernel<<<(WQ_ELEMS + 255) / 256, 256, 0, stream>>>(qkv_w, proj_w, wq, wp);

  hipFuncSetAttribute((const void*)winattn_kernel,
                      hipFuncAttributeMaxDynamicSharedMemorySize, SMEM_BYTES);
  winattn_kernel<<<4096, 256, SMEM_BYTES, stream>>>(x, mask, wq, qkv_b, wp,
                                                    proj_b, out);
}
